// TransegerJointNet_85237920956984
// MI455X (gfx1250) — compile-verified, matrix operand order checked by reading, not run on hardware
//
#include <hip/hip_runtime.h>

#define Bc 2
#define Tc 512
#define Uc 48
#define Dc 512
#define Hc 1024
#define Kc 128

typedef __attribute__((ext_vector_type(16))) _Float16 v16h;
typedef __attribute__((ext_vector_type(8)))  float    v8f;
typedef __attribute__((ext_vector_type(4)))  unsigned v4u;
typedef __attribute__((ext_vector_type(8)))  unsigned v8u;

__device__ __forceinline__ float wave_reduce_sum(float v) {
#pragma unroll
    for (int m = 16; m > 0; m >>= 1) v += __shfl_xor(v, m, 32);
    return v;
}
__device__ __forceinline__ float wave_reduce_max(float v) {
#pragma unroll
    for (int m = 16; m > 0; m >>= 1) v = fmaxf(v, __shfl_xor(v, m, 32));
    return v;
}
__device__ __forceinline__ float fast_tanh(float x) {
    float ax = fabsf(x);
    float e  = __expf(-2.0f * ax);          // v_exp_f32 path (TRANS unit)
    float t  = (1.0f - e) / (1.0f + e);
    return copysignf(t, x);
}

// ---------------------------------------------------------------------------
// img_pre[bt][h] = dot(img_feature[bt,:], W1[h, D:2D])
// ---------------------------------------------------------------------------
__global__ __launch_bounds__(256) void k_img_pre(const float* __restrict__ imgf,
                                                 const float* __restrict__ W1,
                                                 float* __restrict__ img_pre) {
    __shared__ __align__(16) float sx[Dc];
    int bt = blockIdx.x;
    const float* row = imgf + (size_t)bt * Dc;
    for (int i = threadIdx.x; i < Dc; i += 256) sx[i] = row[i];
    __syncthreads();
    int wave = threadIdx.x >> 5, lane = threadIdx.x & 31;
    const float4* sx4 = (const float4*)sx;
    for (int h = wave; h < Hc; h += 8) {
        const float4* w4 = (const float4*)(W1 + (size_t)h * (2 * Dc) + Dc);
        float acc = 0.f;
#pragma unroll
        for (int i = 0; i < (Dc / 4) / 32; ++i) {
            float4 a = sx4[lane + 32 * i];
            float4 b = w4[lane + 32 * i];
            acc += a.x * b.x + a.y * b.y + a.z * b.z + a.w * b.w;
        }
        acc = wave_reduce_sum(acc);
        if (lane == 0) img_pre[(size_t)bt * Hc + h] = acc;
    }
}

// ---------------------------------------------------------------------------
// lab_pre[bu][h] = dot(label_feature[bu,:], W1[h, 0:D]) + b1[h]
// ---------------------------------------------------------------------------
__global__ __launch_bounds__(256) void k_lab_pre(const float* __restrict__ labf,
                                                 const float* __restrict__ W1,
                                                 const float* __restrict__ b1,
                                                 float* __restrict__ lab_pre) {
    __shared__ __align__(16) float sx[Dc];
    int bu = blockIdx.x;
    const float* row = labf + (size_t)bu * Dc;
    for (int i = threadIdx.x; i < Dc; i += 256) sx[i] = row[i];
    __syncthreads();
    int wave = threadIdx.x >> 5, lane = threadIdx.x & 31;
    const float4* sx4 = (const float4*)sx;
    for (int h = wave; h < Hc; h += 8) {
        const float4* w4 = (const float4*)(W1 + (size_t)h * (2 * Dc));
        float acc = 0.f;
#pragma unroll
        for (int i = 0; i < (Dc / 4) / 32; ++i) {
            float4 a = sx4[lane + 32 * i];
            float4 b = w4[lane + 32 * i];
            acc += a.x * b.x + a.y * b.y + a.z * b.z + a.w * b.w;
        }
        acc = wave_reduce_sum(acc);
        if (lane == 0) lab_pre[(size_t)bu * Hc + h] = acc + b1[h];
    }
}

// ---------------------------------------------------------------------------
// W2 f32 -> f16 (row-major [K][H]) — stays L2-resident for the hot loop.
// ---------------------------------------------------------------------------
__global__ __launch_bounds__(256) void k_cvt(const float* __restrict__ W2,
                                             _Float16* __restrict__ W2h, int n) {
    int i = blockIdx.x * 256 + threadIdx.x;
    if (i < n) W2h[i] = (_Float16)W2[i];
}

// ---------------------------------------------------------------------------
// img_seg_score[b,k,t]
// ---------------------------------------------------------------------------
__global__ __launch_bounds__(256) void k_img_seg(const float* __restrict__ imgf,
                                                 const float* __restrict__ conv_w,
                                                 const float* __restrict__ conv_b,
                                                 const float* __restrict__ masks,
                                                 float* __restrict__ out0) {
    __shared__ __align__(16) float sx[Dc];
    int bt = blockIdx.x;
    int b = bt / Tc, t = bt % Tc;
    const float* row = imgf + (size_t)bt * Dc;
    for (int i = threadIdx.x; i < Dc; i += 256) sx[i] = row[i];
    __syncthreads();
    int wave = threadIdx.x >> 5, lane = threadIdx.x & 31;
    float mask = masks[b * Tc + t];
    const float4* sx4 = (const float4*)sx;
    for (int k = wave; k < Kc; k += 8) {
        const float4* w4 = (const float4*)(conv_w + (size_t)k * Dc);
        float acc = 0.f;
#pragma unroll
        for (int i = 0; i < (Dc / 4) / 32; ++i) {
            float4 a = sx4[lane + 32 * i];
            float4 b2_ = w4[lane + 32 * i];
            acc += a.x * b2_.x + a.y * b2_.y + a.z * b2_.z + a.w * b2_.w;
        }
        acc = wave_reduce_sum(acc);
        if (lane == 0) out0[((size_t)b * Kc + k) * Tc + t] = (acc + conv_b[k]) * mask;
    }
}

// ---------------------------------------------------------------------------
// TDM: DMA a 128x32 f16 tile of W2 (row stride H elements) into LDS with
// hardware padding: 16 DWORDs (64B row) + 4 DWORDs pad -> 80B LDS row stride,
// matching LDAB=40 halves. 2D tensor -> 2-SGPR-group form (VADDR2/3 = NULL).
// ---------------------------------------------------------------------------
#define LDAB 40   // padded row stride in halves (80B)
#define SJLD 132  // padded row stride for joint staging

__device__ __forceinline__ void tdm_load_w2(const _Float16* gsrc, unsigned lds_addr) {
    unsigned long long ga = (unsigned long long)(size_t)gsrc;
    v4u g0;
    g0.x = 1u;                                   // count=1, user mode, no gather
    g0.y = lds_addr;                             // D#.lds_addr (bytes)
    g0.z = (unsigned)ga;                         // global_addr[31:0]
    g0.w = ((unsigned)(ga >> 32) & 0x01FFFFFFu)  // global_addr[56:32]
           | 0x80000000u;                        // type=2 ("image") in [127:126]
    v8u g1;
    g1[0] = (1u << 16)      // data_size=1 (2B elements)
          | (1u << 20)      // pad_enable
          | (3u << 22)      // pad_interval: 16 DWORDs (64B) between pads
          | (3u << 25);     // pad_amount: 4 DWORDs (16B)
    g1[1] = (Hc & 0xFFFFu) << 16;     // tensor_dim0[15:0] = 1024 (elements)
    g1[2] = ((unsigned)Kc) << 16;     // tensor_dim0[31:16]=0 | tensor_dim1 = 128
    g1[3] = 32u << 16;                // tensor_dim1[31:16]=0 | tile_dim0 = 32
    g1[4] = (unsigned)Kc;             // tile_dim1 = 128, tile_dim2 = 0
    g1[5] = (unsigned)Hc;             // tensor_dim0_stride[31:0] = 1024
    g1[6] = 0u;                       // stride[47:32]=0 | dim1_stride lo
    g1[7] = 0u;
    asm volatile("tensor_load_to_lds %0, %1" :: "s"(g0), "s"(g1) : "memory");
}

// ---------------------------------------------------------------------------
// Main fused kernel: double-buffered TDM pipeline + tanh A-tile regen + WMMA
// + log_softmax epilogue. One block per (b,t); 8 wave32s.
// ---------------------------------------------------------------------------
__global__ __launch_bounds__(256) void k_joint(const float* __restrict__ img_pre,
                                               const float* __restrict__ lab_pre,
                                               const _Float16* __restrict__ W2h,
                                               const float* __restrict__ b2,
                                               float* __restrict__ outj) {
    __shared__ __align__(16) _Float16 sA[2][Uc * LDAB];   // tanh tiles (dbl buf)
    __shared__ __align__(16) _Float16 sB[2][Kc * LDAB];   // W2 tiles via TDM
    __shared__ __align__(16) float    sJ[Uc * SJLD];      // joint staging

    int bt = blockIdx.x;
    int b = bt / Tc, t = bt % Tc;
    int tid = threadIdx.x;
    int wave = tid >> 5, lane = tid & 31;
    int nl = lane & 15, hi = lane >> 4;
    int n0 = wave * 16;

    const float* imgrow  = img_pre + (size_t)bt * Hc;
    const float* labbase = lab_pre + (size_t)b * Uc * Hc;

    v8f acc[3] = {};

    // Stage A tile (48x32 f16 tanh activations) for chunk at h0 into sA[buf]
    auto stageA = [&](int h0, int buf) {
#pragma unroll
        for (int i = 0; i < 3; ++i) {
            int p = tid + i * 256;         // 0..767
            int u = p >> 4;                // 0..47
            int c = (p & 15) * 2;          // even column 0..30
            float x0 = labbase[(size_t)u * Hc + h0 + c]     + imgrow[h0 + c];
            float x1 = labbase[(size_t)u * Hc + h0 + c + 1] + imgrow[h0 + c + 1];
            _Float16 h2[2] = {(_Float16)fast_tanh(x0), (_Float16)fast_tanh(x1)};
            *(unsigned*)((char*)sA[buf] + u * (LDAB * 2) + c * 2) =
                *(unsigned*)h2;
        }
    };

    // Prologue: kick off chunk 0
    if (wave == 0)
        tdm_load_w2(W2h, (unsigned)(size_t)(void*)&sB[0][0]);
    stageA(0, 0);

    for (int i = 0; i < Hc / 32; ++i) {
        int buf = i & 1;
        if (wave == 0) __builtin_amdgcn_s_wait_tensorcnt(0);  // chunk i in LDS
        __syncthreads();                                      // + sA[buf] visible

        if (i + 1 < Hc / 32) {          // prefetch chunk i+1 into other buffers
            if (wave == 0)
                tdm_load_w2(W2h + (size_t)(i + 1) * 32,
                            (unsigned)(size_t)(void*)&sB[buf ^ 1][0]);
            stageA((i + 1) * 32, buf ^ 1);
        }

        union Frag { uint4 q[2]; v16h v; };
        Frag fb;
        {
            const char* base = (const char*)sB[buf] + (n0 + nl) * (LDAB * 2);
            fb.q[0] = *(const uint4*)(base + hi * 16);        // K 0-7  / 8-15
            fb.q[1] = *(const uint4*)(base + 32 + hi * 16);   // K16-23 / 24-31
        }
#pragma unroll
        for (int m = 0; m < 3; ++m) {
            Frag fa;
            const char* base = (const char*)sA[buf] + (m * 16 + nl) * (LDAB * 2);
            fa.q[0] = *(const uint4*)(base + hi * 16);
            fa.q[1] = *(const uint4*)(base + 32 + hi * 16);
            acc[m] = __builtin_amdgcn_wmma_f32_16x16x32_f16(
                false, fa.v, false, fb.v, (short)0, acc[m], false, false);
        }
    }

    // Epilogue: +b2, stage joint (C layout: M = m*16 + hi*8 + j, N = n0+nl)
    float bias = b2[n0 + nl];
    __syncthreads();
#pragma unroll
    for (int m = 0; m < 3; ++m) {
#pragma unroll
        for (int j = 0; j < 8; ++j) {
            int u = m * 16 + hi * 8 + j;
            sJ[u * SJLD + n0 + nl] = acc[m][j] + bias;
        }
    }
    __syncthreads();

    // log_softmax over K=128 per row u; wave handles rows u = wave, wave+8, ...
    for (int u = wave; u < Uc; u += 8) {
        float x[4];
#pragma unroll
        for (int i = 0; i < 4; ++i) x[i] = sJ[u * SJLD + lane + 32 * i];
        float mx = fmaxf(fmaxf(x[0], x[1]), fmaxf(x[2], x[3]));
        mx = wave_reduce_max(mx);
        float s = 0.f;
#pragma unroll
        for (int i = 0; i < 4; ++i) s += __expf(x[i] - mx);
        s = wave_reduce_sum(s);
        float lse = mx + __logf(s);
        size_t base = (((size_t)b * Uc + u) * Tc + t) * (size_t)Kc;
#pragma unroll
        for (int i = 0; i < 4; ++i) outj[base + lane + 32 * i] = x[i] - lse;
    }
}

// ---------------------------------------------------------------------------
extern "C" void kernel_launch(void* const* d_in, const int* in_sizes, int n_in,
                              void* d_out, int out_size, void* d_ws, size_t ws_size,
                              hipStream_t stream) {
    const float* imgf   = (const float*)d_in[0];   // (B,T,D)
    const float* labf   = (const float*)d_in[1];   // (B,U,D)
    const float* masks  = (const float*)d_in[2];   // (B,1,T)
    const float* W1     = (const float*)d_in[3];   // (H,2D)
    const float* b1     = (const float*)d_in[4];   // (H)
    const float* W2     = (const float*)d_in[5];   // (K,H)
    const float* b2     = (const float*)d_in[6];   // (K)
    const float* conv_w = (const float*)d_in[7];   // (K,D)
    const float* conv_b = (const float*)d_in[8];   // (K)
    (void)in_sizes; (void)n_in; (void)out_size; (void)ws_size;

    float* out       = (float*)d_out;
    float* out_seg   = out;                                   // (B,K,T)
    float* out_joint = out + (size_t)Bc * Kc * Tc;            // (B,U,T,K)

    char* ws = (char*)d_ws;
    float*    img_pre = (float*)ws;                                          // B*T*H f32
    float*    lab_pre = (float*)(ws + (size_t)Bc * Tc * Hc * 4);             // B*U*H f32
    _Float16* W2h     = (_Float16*)(ws + (size_t)Bc * Tc * Hc * 4
                                       + (size_t)Bc * Uc * Hc * 4);          // K*H f16

    k_img_pre<<<Bc * Tc, 256, 0, stream>>>(imgf, W1, img_pre);
    k_lab_pre<<<Bc * Uc, 256, 0, stream>>>(labf, W1, b1, lab_pre);
    k_cvt<<<(Kc * Hc + 255) / 256, 256, 0, stream>>>(W2, W2h, Kc * Hc);
    k_img_seg<<<Bc * Tc, 256, 0, stream>>>(imgf, conv_w, conv_b, masks, out_seg);
    k_joint<<<Bc * Tc, 256, 0, stream>>>(img_pre, lab_pre, W2h, b2, out_joint);
}
